// Reconstructeur_13357348290803
// MI455X (gfx1250) — matrix operand order repaired and checked by reading, not compiled, not run on hardware
//
#include <hip/hip_runtime.h>
#include <cmath>

typedef __attribute__((ext_vector_type(16))) _Float16 v16h;
typedef __attribute__((ext_vector_type(8)))  _Float16 v8h;
typedef __attribute__((ext_vector_type(2)))  _Float16 v2h;
typedef __attribute__((ext_vector_type(8)))  float    v8f;
typedef __attribute__((ext_vector_type(4)))  float    v4f;
typedef __attribute__((ext_vector_type(2)))  float    v2f;

#define N_MLP  16
#define LATENT 1024
#define S0     512
#define S1     256
#define S2     128
#define OUTC   3

// ---------------------------------------------------------------------------
// Prelude 1: c0[m][o] = dot(W0a[m][o][:], x0) + b0a[m][o] + b0b[m][o]
// one wave32 per output row; coalesced 128B strides, shuffle reduction.
// ---------------------------------------------------------------------------
__global__ void gemv_c0_kernel(const float* __restrict__ W0a, const float* __restrict__ x0,
                               const float* __restrict__ b0a, const float* __restrict__ b0b,
                               float* __restrict__ c0) {
  int gt   = blockIdx.x * blockDim.x + threadIdx.x;
  int wid  = gt >> 5;
  int lane = gt & 31;
  if (wid >= N_MLP * S0) return;
  const float* w = W0a + (size_t)wid * LATENT;
  float s = 0.f;
#pragma unroll 4
  for (int k = lane; k < LATENT; k += 32) s = fmaf(w[k], x0[k], s);
#pragma unroll
  for (int off = 16; off > 0; off >>= 1) s += __shfl_down(s, off, 32);
  if (lane == 0) c0[wid] = s + b0a[wid] + b0b[wid];
}

// ---------------------------------------------------------------------------
// Prelude 2: f32 -> f16 weight conversion (grid-stride)
// ---------------------------------------------------------------------------
__global__ void cvt_f32_to_f16(const float* __restrict__ src, _Float16* __restrict__ dst, int n) {
  int stride = gridDim.x * blockDim.x;
  for (int i = blockIdx.x * blockDim.x + threadIdx.x; i < n; i += stride)
    dst[i] = (_Float16)src[i];
}

// ---------------------------------------------------------------------------
// Prelude 3: W3 [16,3,128] -> zero-padded f16 [16,16,128] so layer 3 is WMMA
// ---------------------------------------------------------------------------
__global__ void pad_w3_kernel(const float* __restrict__ W3, _Float16* __restrict__ dst) {
  int i = blockIdx.x * blockDim.x + threadIdx.x;
  if (i >= N_MLP * 16 * S2) return;
  int k  = i % S2;
  int o  = (i / S2) % 16;
  int mm = i / (16 * S2);
  dst[i] = (o < OUTC) ? (_Float16)W3[((size_t)mm * OUTC + o) * S2 + k] : (_Float16)0.f;
}

// ---------------------------------------------------------------------------
// WMMA fragment loaders (CDNA5 16-bit layouts, wave32)
// A (16x32 f16) from LDS row-major [16][ld]:
//   lane<16: elems 0..7 = K k0..k0+7,  8..15 = K k0+16..k0+23   (M = lane)
//   lane>=16: K base +8                                           (M = lane-16)
// B (32x16 f16) from global row-major W[col][ld] (col = output channel):
//   lane<16: elems 0..7 = K k0..k0+7, 8..15 = K k0+16..k0+23 of column lane; lane>=16: K base +8
// ---------------------------------------------------------------------------
__device__ __forceinline__ v16h load_a_lds(const _Float16* h, int ld, int lane, int k0) {
  int mrow = lane & 15;
  int kb   = k0 + ((lane >> 4) << 3);
  const _Float16* p = h + mrow * ld + kb;
  v8h lo = *(const v8h*)p;          // K kb..kb+7
  v8h hi = *(const v8h*)(p + 16);   // K kb+16..kb+23
  return __builtin_shufflevector(lo, hi, 0,1,2,3,4,5,6,7,8,9,10,11,12,13,14,15);
}

__device__ __forceinline__ v16h load_b_glb(const _Float16* __restrict__ W, int ld, int n0,
                                           int lane, int k0) {
  int col = n0 + (lane & 15);
  int kb  = k0 + ((lane >> 4) << 3);
  const _Float16* p = W + (size_t)col * ld + kb;
  v8h lo = *(const v8h*)p;
  v8h hi = *(const v8h*)(p + 16);
  return __builtin_shufflevector(lo, hi, 0,1,2,3,4,5,6,7,8,9,10,11,12,13,14,15);
}

#define WMMA_F16(A, B, C) \
  __builtin_amdgcn_wmma_f32_16x16x32_f16(false, (A), false, (B), (short)0, (C), false, false)

// One GEMM layer with 2x2 register blocking:
//   hout[32][NOUT] = relu(hin[32][K] * W^T + bias)
// Per kt: 2 A frags (LDS) + 2 B frags (global) -> 4 WMMAs
// => 1 ds_load_b128 + 1 global_load_b128 per WMMA (operand-bandwidth balanced)
__device__ __forceinline__ void layer_wmma(const _Float16* hin, int K,
                                           const _Float16* __restrict__ W,
                                           const float* __restrict__ bias,
                                           _Float16* hout, int NOUT, int lane) {
  const int nIdx = lane & 15;
  const int hiM  = (lane >> 4) << 3;
  for (int nt = 0; nt < NOUT; nt += 32) {
    v8f a00 = {}, a01 = {}, a10 = {}, a11 = {};
    for (int kt = 0; kt < K; kt += 32) {
      v16h A0 = load_a_lds(hin,          K, lane, kt);   // rows  0..15
      v16h A1 = load_a_lds(hin + 16 * K, K, lane, kt);   // rows 16..31
      v16h B0 = load_b_glb(W, K, nt,      lane, kt);
      v16h B1 = load_b_glb(W, K, nt + 16, lane, kt);
      a00 = WMMA_F16(A0, B0, a00);
      a10 = WMMA_F16(A1, B0, a10);
      a01 = WMMA_F16(A0, B1, a01);
      a11 = WMMA_F16(A1, B1, a11);
    }
    float bb0 = bias[nt + nIdx];
    float bb1 = bias[nt + 16 + nIdx];
#pragma unroll
    for (int v = 0; v < 8; ++v) {           // D: row M=v+hiM (+16 for tile 1), col N=nIdx
      hout[(v + hiM) * NOUT      + nt      + nIdx] = (_Float16)fmaxf(a00[v] + bb0, 0.f);
      hout[(v + hiM + 16) * NOUT + nt      + nIdx] = (_Float16)fmaxf(a10[v] + bb0, 0.f);
      hout[(v + hiM) * NOUT      + nt + 16 + nIdx] = (_Float16)fmaxf(a01[v] + bb1, 0.f);
      hout[(v + hiM + 16) * NOUT + nt + 16 + nIdx] = (_Float16)fmaxf(a11[v] + bb1, 0.f);
    }
  }
}

// ---------------------------------------------------------------------------
// Fused MLP: 2 waves/block, each wave owns 32 grid points end-to-end.
// LDS per wave: h0 32x512 f16 (32KB) + h1 32x256 f16 (16KB); h2 reuses h0.
// Block = 96KB LDS -> 3 blocks per 320KB WGP.
// blockIdx.y = MLP index, blockIdx.x = grid tile of 64 points.
// ---------------------------------------------------------------------------
__global__ __launch_bounds__(64)
void mlp_fused_kernel(const float* __restrict__ c0,  const float* __restrict__ W0b,
                      const _Float16* __restrict__ W1h, const float* __restrict__ b1,
                      const _Float16* __restrict__ W2h, const float* __restrict__ b2,
                      const _Float16* __restrict__ W3h, const float* __restrict__ b3,
                      float* __restrict__ out, int n_grid, int G) {
  extern __shared__ _Float16 smem[];
  const int lane  = threadIdx.x & 31;
  const int wave  = threadIdx.x >> 5;
  const int m     = blockIdx.y;
  const int gbase = blockIdx.x * 64 + wave * 32;

  _Float16* h0 = smem + (size_t)wave * (32 * S0 + 32 * S1);
  _Float16* h1 = h0 + 32 * S0;
  _Float16* h2 = h0;  // h0 dead after layer 1

  // ---- stage 0: h0[r][o] = relu(c0[o] + gx*W0b[o][0] + gy*W0b[o][1]) --------
  // lane covers output pairs o = j*64 + lane*2 (+0,+1); coefficients kept in
  // registers across all 32 rows; paired f16 results stored as one b32.
  const float step = 2.0f / (float)(n_grid - 1);
  const float* c0m  = c0  + m * S0;
  const float* w0bm = W0b + (size_t)m * S0 * 2;
  v2f ca[8];
  v4f wa[8];
#pragma unroll
  for (int j = 0; j < 8; ++j) {
    int o = j * 64 + lane * 2;
    ca[j] = *(const v2f*)(c0m + o);
    wa[j] = *(const v4f*)(w0bm + o * 2);   // {w[o].x, w[o].y, w[o+1].x, w[o+1].y}
  }
  for (int r = 0; r < 32; ++r) {
    int g = gbase + r; if (g >= G) g = G - 1;
    float gx = -1.0f + step * (float)(g % n_grid);
    float gy = -1.0f + step * (float)(g / n_grid);
#pragma unroll
    for (int j = 0; j < 8; ++j) {
      int o = j * 64 + lane * 2;
      float v0 = fmaf(gx, wa[j].x, fmaf(gy, wa[j].y, ca[j].x));
      float v1 = fmaf(gx, wa[j].z, fmaf(gy, wa[j].w, ca[j].y));
      v2h pk = { (_Float16)fmaxf(v0, 0.f), (_Float16)fmaxf(v1, 0.f) };
      *(v2h*)(h0 + r * S0 + o) = pk;       // 4B-aligned ds_store_b32
    }
  }
  __syncthreads();

  // ---- layer 1: [32x512] x [512x256] ----
  layer_wmma(h0, S0, W1h + (size_t)m * S1 * S0, b1 + m * S1, h1, S1, lane);
  __syncthreads();
  // ---- layer 2: [32x256] x [256x128] ----
  layer_wmma(h1, S1, W2h + (size_t)m * S2 * S1, b2 + m * S2, h2, S2, lane);
  __syncthreads();

  // ---- layer 3: [32x128] x [128x16(pad)] -> tanh -> out[:, 0..2] ----
  {
    const _Float16* W = W3h + (size_t)m * 16 * S2;
    const int nIdx = lane & 15;
    const int hiM  = (lane >> 4) << 3;
    v8f a0 = {}, a1 = {};
    for (int kt = 0; kt < S2; kt += 32) {
      v16h B  = load_b_glb(W, S2, 0, lane, kt);          // shared across M-tiles
      v16h A0 = load_a_lds(h2,           S2, lane, kt);
      v16h A1 = load_a_lds(h2 + 16 * S2, S2, lane, kt);
      a0 = WMMA_F16(A0, B, a0);
      a1 = WMMA_F16(A1, B, a1);
    }
    if (nIdx < OUTC) {
      float bb = b3[m * OUTC + nIdx];
#pragma unroll
      for (int v = 0; v < 8; ++v) {
        int g0 = gbase + hiM + v;
        int g1 = g0 + 16;
        if (g0 < G) out[((size_t)m * G + g0) * OUTC + nIdx] = tanhf(a0[v] + bb);
        if (g1 < G) out[((size_t)m * G + g1) * OUTC + nIdx] = tanhf(a1[v] + bb);
      }
    }
  }
}

// ---------------------------------------------------------------------------
extern "C" void kernel_launch(void* const* d_in, const int* in_sizes, int n_in,
                              void* d_out, int out_size, void* d_ws, size_t ws_size,
                              hipStream_t stream) {
  const float* x0  = (const float*)d_in[0];
  const float* W0a = (const float*)d_in[1];
  const float* b0a = (const float*)d_in[2];
  const float* W0b = (const float*)d_in[3];
  const float* b0b = (const float*)d_in[4];
  const float* W1  = (const float*)d_in[5];
  const float* b1  = (const float*)d_in[6];
  const float* W2  = (const float*)d_in[7];
  const float* b2  = (const float*)d_in[8];
  const float* W3  = (const float*)d_in[9];
  const float* b3  = (const float*)d_in[10];
  (void)in_sizes; (void)n_in; (void)ws_size;

  const int G      = out_size / (N_MLP * OUTC);        // 16384 for n_grid=128
  const int n_grid = (int)(sqrt((double)G) + 0.5);

  // workspace layout (all 32B-aligned)
  char* ws = (char*)d_ws;
  float*    c0  = (float*)ws;                                             // 32 KB
  _Float16* W1h = (_Float16*)(ws + 32768);                                // 4 MiB
  _Float16* W2h = (_Float16*)(ws + 32768 + (size_t)N_MLP * S1 * S0 * 2);  // 1 MiB
  _Float16* W3h = (_Float16*)(ws + 32768 + (size_t)N_MLP * S1 * S0 * 2
                                         + (size_t)N_MLP * S2 * S1 * 2);  // 64 KiB

  // prelude: latent GEMV + weight conversion (tiny vs main kernel)
  gemv_c0_kernel<<<(N_MLP * S0 * 32 + 255) / 256, 256, 0, stream>>>(W0a, x0, b0a, b0b, c0);
  cvt_f32_to_f16<<<2048, 256, 0, stream>>>(W1, W1h, N_MLP * S1 * S0);
  cvt_f32_to_f16<<<1024, 256, 0, stream>>>(W2, W2h, N_MLP * S2 * S1);
  pad_w3_kernel<<<(N_MLP * 16 * S2 + 255) / 256, 256, 0, stream>>>(W3, W3h);

  // fused MLP: 2 waves x 32 rows = 64 grid points per block
  dim3 grid((G + 63) / 64, N_MLP);
  size_t shmem = 2 * (size_t)(32 * S0 + 32 * S1) * sizeof(_Float16);  // 96 KB
  mlp_fused_kernel<<<grid, 64, shmem, stream>>>(c0, W0b, W1h, b1, W2h, b2, W3h, b3,
                                                (float*)d_out, n_grid, G);
}
